// DynamicLocalPoolPointnet_90769838834282
// MI455X (gfx1250) — matrix operand order repaired and checked by reading, not compiled, not run on hardware
//
#include <hip/hip_runtime.h>
#include <hip/hip_bf16.h>
#include <stdint.h>

// ---------------------------------------------------------------------------
// DynamicLocalPoolPointnet for MI455X (gfx1250, wave32)
//   - All N x {128,256} x 128 GEMMs on v_wmma_f32_16x16x32_f16.
//   - Weights pre-transposed to [col][k] f16 so B fragments are contiguous
//     (2x ds_load_b128 per fragment, no scalar packing).
//   - Weight slices staged into LDS by the Tensor Data Mover
//     (tensor_load_to_lds, double-buffered, s_wait_tensorcnt) while WMMA
//     runs on the other buffer.
//   - 64-row block tile: 8 waves x 4 row-subtiles -> 4 WMMAs per B fragment.
// ---------------------------------------------------------------------------

typedef __attribute__((ext_vector_type(16))) _Float16 v16h;
typedef __attribute__((ext_vector_type(8)))  _Float16 v8h;
typedef __attribute__((ext_vector_type(8)))  float    v8f;
typedef __attribute__((ext_vector_type(4)))  unsigned su4;
typedef __attribute__((ext_vector_type(8)))  unsigned su8;

#define NPTS 80000
#define TT   20000
#define BB   4
#define R2C  16384
#define PADC 1.101f   // 1 + PAD + 0.001

// ------------------------- helpers ---------------------------------------

__device__ __forceinline__ unsigned enc_f32(float f) {
  unsigned u = __float_as_uint(f);
  return (u & 0x80000000u) ? ~u : (u | 0x80000000u);
}
__device__ __forceinline__ float dec_f32(unsigned u) {
  if (u == 0u) return 0.0f;                       // "empty cell" sentinel -> 0
  return (u & 0x80000000u) ? __uint_as_float(u & 0x7fffffffu)
                           : __uint_as_float(~u);
}

// ---- TDM: DMA one 32xK-slice of transposed weights (tile 32 x 128 cols,
//      row stride K elements) into LDS at dstLds. 2D descriptor, groups 0+1.
template<int K>
__device__ __forceinline__ void tdm_issue_wslice(const _Float16* Wt, int kk,
                                                 _Float16* dstLds) {
  unsigned lds_off = (unsigned)(uintptr_t)(void*)dstLds;       // low 32 = LDS offset
  unsigned long long ga =
      (unsigned long long)(uintptr_t)(const void*)Wt + (unsigned long long)kk * 2ull;
  su4 g0 = { 1u,                                  // count=1, user mode
             lds_off,                             // lds_addr
             (unsigned)(ga & 0xffffffffu),        // global_addr lo
             (unsigned)((ga >> 32) & 0x01ffffffu) | (2u << 30) }; // addr hi | type=2
  su8 g1 = { 0x00010000u,                         // data_size=1 (2 bytes)
             (32u << 16),                         // tensor_dim0 = 32 (lo16)
             (128u << 16),                        // dim0 hi=0 | tensor_dim1 = 128
             (32u << 16),                         // dim1 hi=0 | tile_dim0 = 32
             128u,                                // tile_dim1 = 128, tile_dim2 = 0
             (unsigned)K,                         // tensor_dim0_stride = K (lo32)
             0u,                                  // stride0 hi | dim1_stride lo
             0u };
  asm volatile("tensor_load_to_lds %0, %1" :: "s"(g0), "s"(g1) : "memory");
}

// ------------------------- WMMA GEMM --------------------------------------
// Y[N x 128] = post( pre(X)[N x K] @ W[K x 128] + bias + add )
// Wt = transposed f16 weights [128][K]. grid = NPTS/64, block = 256 (8 waves).

template<int K, bool PRE, bool POST, bool BIAS, bool ADD, bool O16, bool O32>
__global__ __launch_bounds__(256)
void gemm_wmma(const _Float16* __restrict__ X, int ldx,
               const _Float16* __restrict__ Wt,
               const float*    __restrict__ bias,
               const _Float16* __restrict__ add,
               _Float16* __restrict__ Yh, float* __restrict__ Yf, int ldy) {
  constexpr int S = K / 32;                       // K-steps
  __shared__ _Float16 Xs[64 * K];                 // 64 rows x K activations
  __shared__ _Float16 Wb[2][128 * 32];            // double-buffered W slice [col][k]

  const int tid  = threadIdx.x;
  const int wave = tid >> 5;
  const int lane = tid & 31;
  const int n16  = lane & 15;
  const int hh   = lane >> 4;
  const int row0 = blockIdx.x * 64;
  const int col  = (wave << 4) + n16;             // 8 waves cover 128 cols

  // stage activations as 8-half vectors, pre-activation ReLU fused
  constexpr int RV = K / 8;
  for (int i = tid; i < 64 * RV; i += 256) {
    int r = i / RV, kc = (i - r * RV) * 8;
    v8h v = *(const v8h*)(X + (size_t)(row0 + r) * ldx + kc);
    if (PRE) {
#pragma unroll
      for (int e = 0; e < 8; ++e)
        v[e] = ((float)v[e] > 0.0f) ? v[e] : (_Float16)0.0f;
    }
    *(v8h*)(Xs + r * K + kc) = v;
  }

  if (wave == 0) tdm_issue_wslice<K>(Wt, 0, &Wb[0][0]);

  v8f acc[4] = {{}, {}, {}, {}};

#pragma unroll
  for (int s = 0; s < S; ++s) {
    if (wave == 0) {
      if (s + 1 < S) {                            // prefetch next slice, then
        tdm_issue_wslice<K>(Wt, (s + 1) * 32, &Wb[(s + 1) & 1][0]);
        __builtin_amdgcn_s_wait_tensorcnt(1);     // wait for current slice only
      } else {
        __builtin_amdgcn_s_wait_tensorcnt(0);
      }
    }
    __syncthreads();                              // publish slice s to all waves

    const _Float16* wb = &Wb[s & 1][0];
    // B fragment: per lane 16 contiguous halfs (K = hh*16 + 0..15, fixed col)
    v16h bf;
    *(v8h*)&bf       = *(const v8h*)(wb + col * 32 + hh * 16);
    *((v8h*)&bf + 1) = *(const v8h*)(wb + col * 32 + hh * 16 + 8);

    const int kk = s * 32;
#pragma unroll
    for (int rt = 0; rt < 4; ++rt) {
      // A fragment: row n16, K = hh*16 + 0..15
      const _Float16* xs = Xs + (rt * 16 + n16) * K + kk + hh * 16;
      v16h af;
      *(v8h*)&af       = *(const v8h*)xs;
      *((v8h*)&af + 1) = *(const v8h*)(xs + 8);
      acc[rt] = __builtin_amdgcn_wmma_f32_16x16x32_f16(
          false, af, false, bf, (short)0, acc[rt], false, false);
    }
    __syncthreads();                              // before slice s+2 overwrites
  }

  float bv = BIAS ? bias[col] : 0.0f;
#pragma unroll
  for (int rt = 0; rt < 4; ++rt) {
#pragma unroll
    for (int r = 0; r < 8; ++r) {                 // C/D: row = r + 8*hh
      int row = row0 + rt * 16 + r + (hh << 3);
      float v = acc[rt][r] + bv;
      if (ADD) v += (float)add[(size_t)row * 128 + col];
      if (POST) v = fmaxf(v, 0.0f);
      size_t o = (size_t)row * ldy + col;
      if (O16) Yh[o] = (_Float16)v;
      if (O32) Yf[o] = v;
    }
  }
}

// ------------------------- scalar kernels ---------------------------------

// f32 [nmat][K][128] -> f16 transposed [nmat][128][K]
__global__ void conv_transpose(const float* __restrict__ s,
                               _Float16* __restrict__ d, int K, long total) {
  long i = (long)blockIdx.x * 256 + threadIdx.x;
  if (i >= total) return;
  long per = (long)K * 128;
  int m = (int)(i / per);
  long rem = i - (long)m * per;
  int c = (int)(rem / K), k = (int)(rem - (long)c * K);
  d[i] = (_Float16)s[(long)m * per + (long)k * 128 + c];
}

// K=3 input layers: Y[N x M] = act(p @ W[3 x M] + b)
__global__ void in3_gemm(const float* __restrict__ P,
                         const float* __restrict__ W,
                         const float* __restrict__ b,
                         _Float16* __restrict__ Y, int M, int ldy, int relu) {
  long i = (long)blockIdx.x * 256 + threadIdx.x;
  if (i >= (long)NPTS * M) return;
  int n = (int)(i / M), m = (int)(i - (long)n * M);
  float v = P[n * 3] * W[m] + P[n * 3 + 1] * W[M + m] +
            P[n * 3 + 2] * W[2 * M + m] + b[m];
  if (relu) v = fmaxf(v, 0.0f);
  Y[(size_t)n * ldy + m] = (_Float16)v;
}

// max over T per (batch, channel) into encoded uints
__global__ void rowmax_kernel(const _Float16* __restrict__ H,
                              unsigned* __restrict__ out) {
  int c = threadIdx.x;                  // 128
  int b = blockIdx.x;                   // 4
  int t0 = blockIdx.y * 256;
  int t1 = t0 + 256; if (t1 > TT) t1 = TT;
  float m = -3.4e38f;
  for (int t = t0; t < t1; ++t)
    m = fmaxf(m, (float)H[((size_t)b * TT + t) * 128 + c]);
  atomicMax(&out[b * 128 + c], enc_f32(m));
}

// single block: a=relu(netpl); normals; change_basis -> C_mat; hdim
__global__ void normals_kernel(const unsigned* __restrict__ npl,
                               const float* __restrict__ pp_W,
                               const float* __restrict__ pp_b,
                               const float* __restrict__ ph_W,
                               const float* __restrict__ ph_b,
                               float* __restrict__ Cws,
                               float* __restrict__ Cout,
                               float* __restrict__ hdim) {
  __shared__ float a_sh[BB * 128];
  __shared__ float nrm[12][3];
  int tid = threadIdx.x;
  if (tid < BB * 128) a_sh[tid] = fmaxf(dec_f32(npl[tid]), 0.0f);
  __syncthreads();

  if (tid < 12) {
    int b = tid / 3, l = tid % 3;
    float n0[3];
    for (int k = 0; k < 3; ++k) {
      float s = pp_b[l * 3 + k];
      for (int h = 0; h < 128; ++h)
        s += a_sh[b * 128 + h] * pp_W[(l * 128 + h) * 3 + k];
      n0[k] = s;
    }
    nrm[tid][0] = n0[0]; nrm[tid][1] = n0[1]; nrm[tid][2] = n0[2];

    // ---- change_basis ----
    float nl = sqrtf(n0[0] * n0[0] + n0[1] * n0[1] + n0[2] * n0[2]);
    float nx = n0[0] / nl + 1e-4f, ny = n0[1] / nl + 1e-4f, nz = n0[2] / nl + 1e-4f;
    float v0 = -ny, v1 = nx, v2 = 0.0f;                // v = cross(z, n)
    float Km[3][3] = {{0.0f, -v2, v1}, {v2, 0.0f, -v0}, {-v1, v0, 0.0f}};
    float KK[3][3];
    for (int i = 0; i < 3; ++i)
      for (int j = 0; j < 3; ++j)
        KK[i][j] = Km[i][0] * Km[0][j] + Km[i][1] * Km[1][j] + Km[i][2] * Km[2][j];
    float scl = (1.0f - nz) / (v0 * v0 + v1 * v1 + v2 * v2);
    float R[3][3];
    for (int i = 0; i < 3; ++i)
      for (int j = 0; j < 3; ++j)
        R[i][j] = (i == j ? 1.0f : 0.0f) + Km[i][j] + KK[i][j] * scl;
    float det = R[0][0] * (R[1][1] * R[2][2] - R[1][2] * R[2][1]) -
                R[0][1] * (R[1][0] * R[2][2] - R[1][2] * R[2][0]) +
                R[0][2] * (R[1][0] * R[2][1] - R[1][1] * R[2][0]);
    float id = 1.0f / det;
    float Ci[3][3];
    Ci[0][0] =  (R[1][1] * R[2][2] - R[1][2] * R[2][1]) * id;
    Ci[0][1] = -(R[0][1] * R[2][2] - R[0][2] * R[2][1]) * id;
    Ci[0][2] =  (R[0][1] * R[1][2] - R[0][2] * R[1][1]) * id;
    Ci[1][0] = -(R[1][0] * R[2][2] - R[1][2] * R[2][0]) * id;
    Ci[1][1] =  (R[0][0] * R[2][2] - R[0][2] * R[2][0]) * id;
    Ci[1][2] = -(R[0][0] * R[1][2] - R[0][2] * R[1][0]) * id;
    Ci[2][0] =  (R[1][0] * R[2][1] - R[1][1] * R[2][0]) * id;
    Ci[2][1] = -(R[0][0] * R[2][1] - R[0][1] * R[2][0]) * id;
    Ci[2][2] =  (R[0][0] * R[1][1] - R[0][1] * R[1][0]) * id;
    float sbx = 0.0f, sbx2 = 0.0f, sby = 0.0f, sby2 = 0.0f;
    for (int i = 0; i < 3; ++i) {
      float bx = fabsf(R[i][0]), by = fabsf(R[i][1]);
      sbx += bx; sbx2 += bx * bx; sby += by; sby2 += by * by;
    }
    float cc = fmaxf(sbx / sqrtf(sbx2), sby / sqrtf(sby2));

    float* dst = Cws + tid * 12;
    float* dso = Cout + tid * 12;
    for (int i = 0; i < 3; ++i)
      for (int j = 0; j < 3; ++j) { dst[i*3+j] = Ci[i][j]; dso[i*3+j] = Ci[i][j]; }
    for (int j = 0; j < 3; ++j) { dst[9 + j] = cc; dso[9 + j] = cc; }
  }
  __syncthreads();

  for (int i = tid; i < 12 * 128; i += blockDim.x) {
    int pl = i >> 7, c = i & 127;
    int l = pl % 3;
    float s = ph_b[l * 128 + c];
    for (int k = 0; k < 3; ++k)
      s += nrm[pl][k] * ph_W[(l * 3 + k) * 128 + c];
    hdim[i] = s;
  }
}

__global__ void plane_index_kernel(const float* __restrict__ P,
                                   const float* __restrict__ Cws,
                                   int* __restrict__ idx) {
  long i = (long)blockIdx.x * 256 + threadIdx.x;
  if (i >= 3L * NPTS) return;
  int l = (int)(i / NPTS);
  int n = (int)(i - (long)l * NPTS);
  int b = n / TT;
  const float* C = Cws + (b * 3 + l) * 12;
  float cc = C[9];
  float px = P[n * 3], py = P[n * 3 + 1], pz = P[n * 3 + 2];
  float q0 = (C[0] * px + C[1] * py + C[2] * pz) / cc;
  float q1 = (C[3] * px + C[4] * py + C[5] * pz) / cc;
  float x = fminf(fmaxf(q0 / PADC + 0.5f, 0.0f), 1.0f - 1e-5f);
  float y = fminf(fmaxf(q1 / PADC + 0.5f, 0.0f), 1.0f - 1e-5f);
  idx[(long)l * NPTS + n] = (int)(x * 128.0f) + 128 * (int)(y * 128.0f);
}

__global__ void scatter_max_kernel(const _Float16* __restrict__ net, int ldn,
                                   const int* __restrict__ idx,
                                   unsigned* __restrict__ scat) {
  long i = (long)blockIdx.x * 256 + threadIdx.x;
  if (i >= (long)NPTS * 128) return;
  int n = (int)(i >> 7), c = (int)(i & 127);
  int b = n / TT;
  float v = (float)net[(size_t)n * ldn + c];
  atomicMax(&scat[((size_t)b * R2C + idx[n]) * 128 + c], enc_f32(v));
}

__global__ void gather_acc_kernel(const unsigned* __restrict__ scat,
                                  const int* __restrict__ idx,
                                  float* __restrict__ pool) {
  long i = (long)blockIdx.x * 256 + threadIdx.x;
  if (i >= (long)NPTS * 128) return;
  int n = (int)(i >> 7), c = (int)(i & 127);
  int b = n / TT;
  pool[i] += dec_f32(scat[((size_t)b * R2C + idx[n]) * 128 + c]);
}

__global__ void pack_pool_kernel(const float* __restrict__ pool,
                                 _Float16* __restrict__ dst) {  // dst = buf+128
  long i = (long)blockIdx.x * 256 + threadIdx.x;
  if (i >= (long)NPTS * 128) return;
  int n = (int)(i >> 7), c = (int)(i & 127);
  dst[(size_t)n * 256 + c] = (_Float16)pool[i];
}

__global__ void scatter_sum_kernel(const float* __restrict__ cf,
                                   const float* __restrict__ hdim, int l,
                                   const int* __restrict__ idx,
                                   float* __restrict__ sum,
                                   float* __restrict__ cnt) {
  long i = (long)blockIdx.x * 256 + threadIdx.x;
  if (i >= (long)NPTS * 128) return;
  int n = (int)(i >> 7), c = (int)(i & 127);
  int b = n / TT;
  float v = cf[i] + hdim[(b * 3 + l) * 128 + c];
  atomicAdd(&sum[((size_t)b * R2C + idx[n]) * 128 + c], v);
  if (c == 0) atomicAdd(&cnt[(size_t)b * R2C + idx[n]], 1.0f);
}

__global__ void finalize_kernel(const float* __restrict__ sum,
                                const float* __restrict__ cnt,
                                float* __restrict__ out) {  // out offset by l
  long i = (long)blockIdx.x * 256 + threadIdx.x;
  if (i >= (long)BB * 128 * R2C) return;
  int r = (int)(i & (R2C - 1));
  long t = i >> 14;
  int c = (int)(t & 127);
  int b = (int)(t >> 7);
  float s = sum[((size_t)b * R2C + r) * 128 + c];
  float k = fmaxf(cnt[(size_t)b * R2C + r], 1.0f);
  out[((size_t)(b * 128 + c)) * R2C + r] = s / k;
}

// ------------------------- launcher ----------------------------------------

static inline int cdiv(long a, int b) { return (int)((a + b - 1) / b); }

template<int K, bool PRE, bool POST, bool BIAS, bool ADD, bool O16, bool O32>
static void launch_gemm(const _Float16* X, int ldx, const _Float16* Wt,
                        const float* bias, const _Float16* add,
                        _Float16* Yh, float* Yf, int ldy, hipStream_t s) {
  gemm_wmma<K, PRE, POST, BIAS, ADD, O16, O32>
      <<<NPTS / 64, 256, 0, s>>>(X, ldx, Wt, bias, add, Yh, Yf, ldy);
}

extern "C" void kernel_launch(void* const* d_in, const int* in_sizes, int n_in,
                              void* d_out, int out_size, void* d_ws, size_t ws_size,
                              hipStream_t stream) {
  const float* p      = (const float*)d_in[0];
  const float* W_pos  = (const float*)d_in[1];
  const float* b_pos  = (const float*)d_in[2];
  const float* fc0_W  = (const float*)d_in[3];
  const float* fc0_b  = (const float*)d_in[4];
  const float* fc1_W  = (const float*)d_in[5];
  const float* fc1_b  = (const float*)d_in[6];
  const float* sc_W   = (const float*)d_in[7];
  const float* W_c    = (const float*)d_in[8];
  const float* b_c    = (const float*)d_in[9];
  const float* pn_W1  = (const float*)d_in[10];
  const float* pn_b1  = (const float*)d_in[11];
  const float* pn_W2  = (const float*)d_in[12];
  const float* pn_b2  = (const float*)d_in[13];
  const float* pn_W3  = (const float*)d_in[14];
  const float* pn_b3  = (const float*)d_in[15];
  const float* pp_W   = (const float*)d_in[16];
  const float* pp_b   = (const float*)d_in[17];
  const float* ph_W   = (const float*)d_in[18];
  const float* ph_b   = (const float*)d_in[19];

  float* out = (float*)d_out;
  const size_t FEAS = (size_t)3 * BB * 128 * R2C;   // C_mat follows feas

  // ---- workspace carve ----
  char* base = (char*)d_ws;
  size_t off = 0;
  auto carve = [&](size_t bytes) -> char* {
    char* r = base + off;
    off += (bytes + 255) & ~(size_t)255;
    return r;
  };
  _Float16* fc0h  = (_Float16*)carve((size_t)5 * 256 * 128 * 2);  // transposed
  _Float16* fc1h  = (_Float16*)carve((size_t)5 * 128 * 128 * 2);
  _Float16* sch   = (_Float16*)carve((size_t)5 * 256 * 128 * 2);
  _Float16* Wch   = (_Float16*)carve((size_t)128 * 128 * 2);
  _Float16* pnW2h = (_Float16*)carve((size_t)128 * 128 * 2);
  _Float16* pnW3h = (_Float16*)carve((size_t)128 * 128 * 2);
  _Float16* XB    = (_Float16*)carve((size_t)NPTS * 256 * 2);
  _Float16* YB    = (_Float16*)carve((size_t)NPTS * 256 * 2);
  _Float16* HA    = (_Float16*)carve((size_t)NPTS * 128 * 2);
  _Float16* HB    = (_Float16*)carve((size_t)NPTS * 128 * 2);
  float*    POOL  = (float*)   carve((size_t)NPTS * 128 * 4);     // reused as CF
  unsigned* SCAT  = (unsigned*)carve((size_t)BB * R2C * 128 * 4); // reused as SUM
  float*    CNT   = (float*)   carve((size_t)BB * R2C * 4);
  unsigned* NPL   = (unsigned*)carve((size_t)BB * 128 * 4);
  float*    HD    = (float*)   carve((size_t)12 * 128 * 4);
  float*    CM    = (float*)   carve((size_t)144 * 4);
  int*      IDX   = (int*)     carve((size_t)3 * NPTS * 4);
  (void)ws_size; (void)in_sizes; (void)n_in; (void)out_size;

  // ---- weight conversion: f32 [K][128] -> f16 transposed [128][K] ----
  conv_transpose<<<cdiv(5L*256*128,256),256,0,stream>>>(fc0_W, fc0h, 256, 5L*256*128);
  conv_transpose<<<cdiv(5L*128*128,256),256,0,stream>>>(fc1_W, fc1h, 128, 5L*128*128);
  conv_transpose<<<cdiv(5L*256*128,256),256,0,stream>>>(sc_W,  sch,  256, 5L*256*128);
  conv_transpose<<<cdiv(128L*128,256),256,0,stream>>>(W_c,   Wch,   128, 128L*128);
  conv_transpose<<<cdiv(128L*128,256),256,0,stream>>>(pn_W2, pnW2h, 128, 128L*128);
  conv_transpose<<<cdiv(128L*128,256),256,0,stream>>>(pn_W3, pnW3h, 128, 128L*128);

  // ---- positional encoding: net0 = p @ W_pos + b_pos -> XB (ld 256) ----
  in3_gemm<<<cdiv((long)NPTS*256,256),256,0,stream>>>(p, W_pos, b_pos, XB, 256, 256, 0);

  // ---- pointnet: h1 -> h2 -> h3, max over T ----
  in3_gemm<<<cdiv((long)NPTS*128,256),256,0,stream>>>(p, pn_W1, pn_b1, HA, 128, 128, 1);
  launch_gemm<128,0,1,1,0,1,0>(HA, 128, pnW2h, pn_b2, nullptr, HB, nullptr, 128, stream);
  launch_gemm<128,0,0,1,0,1,0>(HB, 128, pnW3h, pn_b3, nullptr, HA, nullptr, 128, stream);
  hipMemsetAsync(NPL, 0, (size_t)BB * 128 * 4, stream);
  {
    dim3 g(BB, cdiv(TT, 256));
    rowmax_kernel<<<g, 128, 0, stream>>>(HA, NPL);
  }

  // ---- normals / change_basis / hdim (tiny) ----
  normals_kernel<<<1, 512, 0, stream>>>(NPL, pp_W, pp_b, ph_W, ph_b,
                                        CM, out + FEAS, HD);
  plane_index_kernel<<<cdiv(3L*NPTS,256),256,0,stream>>>(p, CM, IDX);

  // ---- residual blocks with inter-block 3-channel scatter-max pooling ----
  _Float16* cur = XB;
  _Float16* nxt = YB;
  const long NE = (long)NPTS * 128;
  for (int i = 0; i < 5; ++i) {
    // h = relu(x) @ W0 + b0
    launch_gemm<256,1,0,1,0,1,0>(cur, 256, fc0h + (size_t)i*256*128,
                                 fc0_b + i*128, nullptr, HA, nullptr, 128, stream);
    // dx = relu(h) @ W1 + b1
    launch_gemm<128,1,0,1,0,1,0>(HA, 128, fc1h + (size_t)i*128*128,
                                 fc1_b + i*128, nullptr, HB, nullptr, 128, stream);
    // net = x @ Ws + dx  -> nxt cols 0..127 (ld 256)
    launch_gemm<256,0,0,0,1,1,0>(cur, 256, sch + (size_t)i*256*128,
                                 nullptr, HB, nxt, nullptr, 256, stream);
    if (i < 4) {
      hipMemsetAsync(POOL, 0, (size_t)NPTS * 128 * 4, stream);
      for (int l = 0; l < 3; ++l) {
        hipMemsetAsync(SCAT, 0, (size_t)BB * R2C * 128 * 4, stream);
        scatter_max_kernel<<<cdiv(NE,256),256,0,stream>>>(nxt, 256,
                                                          IDX + (long)l*NPTS, SCAT);
        gather_acc_kernel<<<cdiv(NE,256),256,0,stream>>>(SCAT,
                                                         IDX + (long)l*NPTS, POOL);
      }
      pack_pool_kernel<<<cdiv(NE,256),256,0,stream>>>(POOL, nxt + 128);
    }
    _Float16* t = cur; cur = nxt; nxt = t;
  }

  // ---- c = net @ W_c + b_c (fp32 out, reuse POOL as CF) ----
  float* CF = POOL;
  launch_gemm<128,0,0,1,0,0,1>(cur, 256, Wch, b_c, nullptr,
                               nullptr, CF, 128, stream);

  // ---- per-channel scatter-mean into output feature planes ----
  for (int l = 0; l < 3; ++l) {
    hipMemsetAsync(SCAT, 0, (size_t)BB * R2C * 128 * 4, stream);
    hipMemsetAsync(CNT,  0, (size_t)BB * R2C * 4, stream);
    scatter_sum_kernel<<<cdiv(NE,256),256,0,stream>>>(POOL, HD, l,
                                                      IDX + (long)l*NPTS,
                                                      (float*)SCAT, CNT);
    finalize_kernel<<<cdiv((long)BB*128*R2C,256),256,0,stream>>>(
        (const float*)SCAT, CNT, out + (size_t)l * BB * 128 * R2C);
  }
}
